// HaloAttn_79096117723730
// MI455X (gfx1250) — matrix operand order repaired and checked by reading, not compiled, not run on hardware
//
#include <hip/hip_runtime.h>
#include <cstddef>
#include <cstdint>

typedef __attribute__((ext_vector_type(16))) __bf16 v16bf;
typedef __attribute__((ext_vector_type(8)))  __bf16 v8bf;
typedef __attribute__((ext_vector_type(8)))  float  v8f;

#define AS1 __attribute__((address_space(1)))
#define AS3 __attribute__((address_space(3)))

#if defined(__HIP_DEVICE_COMPILE__) && defined(__has_builtin)
#if __has_builtin(__builtin_amdgcn_global_load_async_to_lds_b32)
#define HAVE_ASYNC_LDS 1
#endif
#if __has_builtin(__builtin_amdgcn_s_wait_asynccnt)
#define HAVE_WAIT_ASYNC_BUILTIN 1
#endif
#endif

#define HALO   3
#define WIN    14          // 8 + 2*3
#define NPOS   196         // WIN*WIN
#define NPAD   208         // 13 * 16 (S N-dim padding)
#define KPAD   224         // 7 * 32  (O K-dim padding)
#define DH     32
#define DIMC   256
#define PADW   134         // 128 + 2*HALO
#define NBLK   256         // 16*16 blocks per image
#define HWPIX  16384       // 128*128
#define SCALE_QK 0.17677669529663689f

// ---- WMMA operand helpers (contiguous k order used by this file, same for A and B) ----
// A-matrix (16x32 bf16): lane L<16 = row L, element j <-> K = 16*laneH + j
__device__ __forceinline__ v16bf load_a_rowbf(const __bf16* row, int laneH) {
  v8bf c0 = *(const v8bf*)(row + laneH * 16);
  v8bf c1 = *(const v8bf*)(row + laneH * 16 + 8);
  v16bf r;
#pragma unroll
  for (int j = 0; j < 8; ++j) { r[j] = c0[j]; r[j + 8] = c1[j]; }
  return r;
}
// B-matrix (32x16 bf16): lane L = column (L&15), element j <-> K = 16*laneH + j
// -> a single contiguous 32B chunk per lane.

__device__ __forceinline__ v8f wmma_bf16(v16bf a, v16bf b, v8f c) {
  return __builtin_amdgcn_wmma_f32_16x16x32_bf16(false, a, false, b, (short)0, c,
                                                 false, false);
}

__device__ __forceinline__ unsigned packbf2(float a, float b) {
  __bf16 x = (__bf16)a, y = (__bf16)b;
  unsigned short ux, uy;
  __builtin_memcpy(&ux, &x, 2);
  __builtin_memcpy(&uy, &y, 2);
  return (unsigned)ux | ((unsigned)uy << 16);
}

// =====================================================================
// Kernel 1: fused q / kv projection GEMM (768x256) x (256 x B*H*W)
//   grid.x = B(4) * Mtiles(12) * Hrows(128), block = 256 threads = 8 waves
//   WG tile: M=64 out-channels x N=128 pixels (one image row)
// =====================================================================
__global__ __launch_bounds__(256) void halo_proj_kernel(
    const float* __restrict__ x, const float* __restrict__ w_q,
    const float* __restrict__ w_kv, __bf16* __restrict__ qout,
    __bf16* __restrict__ kvp) {
  int bid = blockIdx.x;
  const int hrow = bid & 127; bid >>= 7;
  const int mt = bid % 12;
  const int b  = bid / 12;
  const int Mbase = mt * 64;

  const int tid  = threadIdx.x;
  const int wave = tid >> 5;
  const int lane = tid & 31;
  const int laneN = lane & 15, laneH = lane >> 4;
  const int wm = wave >> 2;  // 0..1  (M sub-tile of 32)
  const int wn = wave & 3;   // 0..3  (N sub-tile of 32)

  __shared__ __align__(32) __bf16 xs[128][32];  // x tile, pixel-major (B layout)

  const v8f zero8 = {0.f, 0.f, 0.f, 0.f, 0.f, 0.f, 0.f, 0.f};
  v8f acc[2][2];
#pragma unroll
  for (int mi = 0; mi < 2; ++mi)
#pragma unroll
    for (int ni = 0; ni < 2; ++ni) acc[mi][ni] = zero8;

  const float* xrow = x + (size_t)b * DIMC * HWPIX + (size_t)hrow * 128;

  // staging role: thread handles a k-pair (packed bf16x2 dword LDS stores)
  const int kp = tid >> 4;        // 0..15 -> channels 2kp, 2kp+1
  const int p0 = (tid & 15) * 8;  // 0..120

#pragma unroll
  for (int kc = 0; kc < 8; ++kc) {
    const int k0 = kc * 32;
    __syncthreads();
    {
      const float4* s0 =
          (const float4*)(xrow + (size_t)(k0 + 2 * kp) * HWPIX + p0);
      const float4* s1 =
          (const float4*)(xrow + (size_t)(k0 + 2 * kp + 1) * HWPIX + p0);
      float4 a0 = s0[0], a1 = s0[1], b0 = s1[0], b1 = s1[1];
      if (kc < 7) {  // prefetch next K-chunk (global_prefetch_b8)
        __builtin_prefetch((const char*)s0 + (size_t)32 * HWPIX * 4, 0, 1);
        __builtin_prefetch((const char*)s1 + (size_t)32 * HWPIX * 4, 0, 1);
      }
      float f0[8] = {a0.x, a0.y, a0.z, a0.w, a1.x, a1.y, a1.z, a1.w};
      float f1[8] = {b0.x, b0.y, b0.z, b0.w, b1.x, b1.y, b1.z, b1.w};
#pragma unroll
      for (int j = 0; j < 8; ++j)
        *(unsigned*)&xs[p0 + j][2 * kp] = packbf2(f0[j], f1[j]);
    }
    __syncthreads();

    v16bf a[2], bb[2];
#pragma unroll
    for (int mi = 0; mi < 2; ++mi) {
      const int o = Mbase + wm * 32 + mi * 16 + laneN;
      const float* wr = (o < 256) ? (w_q + (size_t)o * DIMC + k0)
                                  : (w_kv + (size_t)(o - 256) * DIMC + k0);
      v16bf t;
#pragma unroll
      for (int j = 0; j < 16; ++j)
        t[j] = (__bf16)wr[16 * laneH + j];
      a[mi] = t;
    }
#pragma unroll
    for (int ni = 0; ni < 2; ++ni) {
      const int p = wn * 32 + ni * 16 + laneN;
      bb[ni] = *(const v16bf*)&xs[p][laneH * 16];
    }
#pragma unroll
    for (int mi = 0; mi < 2; ++mi)
#pragma unroll
      for (int ni = 0; ni < 2; ++ni)
        acc[mi][ni] = wmma_bf16(a[mi], bb[ni], acc[mi][ni]);
  }

  // scatter results: q -> [b*8+head][blk][qpos][32] (paired dword stores),
  //                  kv -> padded bf16 planes
  const int by = hrow >> 3, ry = hrow & 7;
#pragma unroll
  for (int mi = 0; mi < 2; ++mi)
#pragma unroll
    for (int ni = 0; ni < 2; ++ni) {
      const int wcol = wn * 32 + ni * 16 + laneN;  // W coordinate 0..127
      const int bx = wcol >> 3, rx = wcol & 7;
      const int nbI = by * 16 + bx, qpos = ry * 8 + rx;
#pragma unroll
      for (int i = 0; i < 8; i += 2) {
        const int o = Mbase + wm * 32 + mi * 16 + i + 8 * laneH;  // even
        if (o < 256) {  // pair (o, o+1) stays within q, d and d+1 adjacent
          const int head = o >> 5, d = o & 31;
          __bf16* p = &qout[(((size_t)(b * 8 + head) * NBLK + nbI) * 64 + qpos) *
                                DH + d];
          *(unsigned*)p = packbf2(acc[mi][ni][i], acc[mi][ni][i + 1]);
        } else {
          const int c = o - 256;
          __bf16* p = &kvp[(((size_t)b * 512 + c) * PADW + (HALO + hrow)) *
                               PADW + (HALO + wcol)];
          p[0] = (__bf16)acc[mi][ni][i];
          p[(size_t)PADW * PADW] = (__bf16)acc[mi][ni][i + 1];
        }
      }
    }
}

// =====================================================================
// Kernel 2: halo attention. One WG (4 waves, 128 thr) per (b, head, block).
//   grid.x = 32 * 256 = 8192
// =====================================================================
__global__ __launch_bounds__(128) void halo_attn_kernel(
    const __bf16* __restrict__ qout, const __bf16* __restrict__ kvp,
    const float* __restrict__ hrel, const float* __restrict__ wrel,
    float* __restrict__ out) {
  int bid = blockIdx.x;
  const int nbI = bid & 255;
  const int bh  = bid >> 8;        // b*8 + head
  const int b = bh >> 3, head = bh & 7;
  const int by = nbI >> 4, bx = nbI & 15;

  const int tid = threadIdx.x;
  const int wave = tid >> 5, lane = tid & 31;
  const int laneN = lane & 15, laneH = lane >> 4;
  const int mbase = wave * 16;  // this wave owns q rows [mbase, mbase+16)

  __shared__ __align__(32) __bf16 k_lds[NPAD][DH];   // pixel-major (B layout for S)
  __shared__ __align__(32) __bf16 v_lds[DH][KPAD];   // channel-major (B layout for O)
  __shared__ __align__(32) __bf16 p_lds[64][KPAD];   // probs, row-major (A layout)
  __shared__ __align__(16) float  rh_lds[64][32];
  __shared__ __align__(16) float  rw_lds[64][32];
  __shared__ __align__(16) float  o_lds[DH][64];

  // all window-row base offsets below are even elements -> 4B aligned
  const __bf16* kbase =
      kvp + (((size_t)b * 512 + head * 64) * PADW + by * 8) * PADW + bx * 8;

  // ---- Phase 0a: v window (no transpose): global -> LDS, async if available
  for (int task = tid; task < 32 * WIN; task += 128) {
    const int c = task / WIN, wy = task % WIN;
    const __bf16* src = kbase + ((size_t)(32 + c) * PADW + wy) * PADW;
#ifdef HAVE_ASYNC_LDS
    AS3 char* dst = (AS3 char*)&v_lds[c][wy * WIN];
#pragma unroll
    for (int j = 0; j < 7; ++j)
      __builtin_amdgcn_global_load_async_to_lds_b32(
          (AS1 int*)((const char*)src + 4 * j), (AS3 int*)(dst + 4 * j), 0, 0);
#else
    const unsigned* s32 = (const unsigned*)src;
    unsigned* d32 = (unsigned*)&v_lds[c][wy * WIN];
#pragma unroll
    for (int j = 0; j < 7; ++j) d32[j] = s32[j];
#endif
  }
  for (int idx = tid; idx < 32 * (KPAD - NPOS) / 2; idx += 128) {  // zero v tail
    const int c = idx / ((KPAD - NPOS) / 2), j = idx % ((KPAD - NPOS) / 2);
    *(unsigned*)&v_lds[c][NPOS + 2 * j] = 0u;
  }

  // ---- Phase 0b: k window with transpose (dword loads, packed dword stores)
  for (int task = tid; task < 16 * WIN; task += 128) {
    const int cp = task / WIN, wy = task % WIN;  // channel pair 2cp, 2cp+1
    const unsigned* r0 =
        (const unsigned*)(kbase + ((size_t)(2 * cp) * PADW + wy) * PADW);
    const unsigned* r1 =
        (const unsigned*)(kbase + ((size_t)(2 * cp + 1) * PADW + wy) * PADW);
    unsigned w0[7], w1[7];
#pragma unroll
    for (int j = 0; j < 7; ++j) { w0[j] = r0[j]; w1[j] = r1[j]; }
#pragma unroll
    for (int pos = 0; pos < WIN; ++pos) {
      const int jw = pos >> 1, sh = (pos & 1) * 16;
      const unsigned h0 = (w0[jw] >> sh) & 0xffffu;
      const unsigned h1 = (w1[jw] >> sh) & 0xffffu;
      *(unsigned*)&k_lds[wy * WIN + pos][2 * cp] = h0 | (h1 << 16);
    }
  }

  // ---- per-wave q A-tile (reused for rel-bias + S) ----
  const __bf16* qbase = qout + ((size_t)bh * NBLK + nbI) * 64 * DH;
  const v16bf a_q = load_a_rowbf(qbase + (size_t)(mbase + laneN) * DH, laneH);

  // ---- rel-position logits: rh = q . hrel^T, rw = q . wrel^T (27 padded->32)
  const v8f zero8 = {0.f, 0.f, 0.f, 0.f, 0.f, 0.f, 0.f, 0.f};
#pragma unroll
  for (int rsel = 0; rsel < 2; ++rsel) {
    const float* R = rsel ? wrel : hrel;
    float(*dst)[32] = rsel ? rw_lds : rh_lds;
#pragma unroll
    for (int nt = 0; nt < 2; ++nt) {
      int col = nt * 16 + laneN;
      if (col > 26) col = 26;  // clamp (cols 27..31 never read back)
      const float* rp = R + (size_t)col * DH + laneH * 16;
      v16bf bt;
#pragma unroll
      for (int j = 0; j < 16; ++j) bt[j] = (__bf16)rp[j];
      const v8f c = wmma_bf16(a_q, bt, zero8);
#pragma unroll
      for (int i = 0; i < 8; ++i)
        dst[mbase + i + 8 * laneH][nt * 16 + laneN] = c[i];
    }
  }

#ifdef HAVE_ASYNC_LDS
#ifdef HAVE_WAIT_ASYNC_BUILTIN
  __builtin_amdgcn_s_wait_asynccnt(0);
#else
  asm volatile("s_wait_asynccnt 0" ::: "memory");
#endif
#endif
  __syncthreads();  // k_lds / v_lds ready

  // ---- Phase 1: S = q . k^T  (13 N-tiles of 16, K=32 in one WMMA each) ----
  v8f s[13];
#pragma unroll
  for (int t = 0; t < 13; ++t) {
    const v16bf bk = *(const v16bf*)&k_lds[t * 16 + laneN][laneH * 16];
    s[t] = wmma_bf16(a_q, bk, zero8);
  }

  // scale + decomposed rel bias + mask (bias index = 13 + k - q, in [6,26])
#pragma unroll
  for (int t = 0; t < 13; ++t) {
    const int n = t * 16 + laneN;
    if (n < NPOS) {
      const int ky = n / WIN, kx = n % WIN;
#pragma unroll
      for (int i = 0; i < 8; ++i) {
        const int m = mbase + i + 8 * laneH;
        const int qy = m >> 3, qx = m & 7;
        s[t][i] = s[t][i] * SCALE_QK + rh_lds[m][13 + ky - qy] +
                  rw_lds[m][13 + kx - qx];
      }
    } else {
#pragma unroll
      for (int i = 0; i < 8; ++i) s[t][i] = -1e30f;
    }
  }

  // ---- softmax over 196 (row = fixed (i, laneH); reduce 13 regs x 16 lanes)
#pragma unroll
  for (int i = 0; i < 8; ++i) {
    float mx = -1e30f;
#pragma unroll
    for (int t = 0; t < 13; ++t) mx = fmaxf(mx, s[t][i]);
    mx = fmaxf(mx, __shfl_xor(mx, 1, 32));
    mx = fmaxf(mx, __shfl_xor(mx, 2, 32));
    mx = fmaxf(mx, __shfl_xor(mx, 4, 32));
    mx = fmaxf(mx, __shfl_xor(mx, 8, 32));
    float sum = 0.f;
#pragma unroll
    for (int t = 0; t < 13; ++t) {
      const float e = __expf(s[t][i] - mx);
      s[t][i] = e;
      sum += e;
    }
    sum += __shfl_xor(sum, 1, 32);
    sum += __shfl_xor(sum, 2, 32);
    sum += __shfl_xor(sum, 4, 32);
    sum += __shfl_xor(sum, 8, 32);
    const float inv = 1.0f / sum;
    const int m = mbase + i + 8 * laneH;
#pragma unroll
    for (int t = 0; t < 13; ++t)
      p_lds[m][t * 16 + laneN] = (__bf16)(s[t][i] * inv);
    p_lds[m][NPAD + laneN] = (__bf16)0.0f;  // zero cols 208..223
  }

  // ---- Phase 2: O = P . V (K padded to 224; per-wave private rows) ----
  v8f o0 = zero8, o1 = zero8;
#pragma unroll
  for (int kc = 0; kc < 7; ++kc) {
    const v16bf ap = load_a_rowbf(&p_lds[mbase + laneN][kc * 32], laneH);
    const v16bf bv0 = *(const v16bf*)&v_lds[laneN][kc * 32 + laneH * 16];
    const v16bf bv1 = *(const v16bf*)&v_lds[16 + laneN][kc * 32 + laneH * 16];
    o0 = wmma_bf16(ap, bv0, o0);
    o1 = wmma_bf16(ap, bv1, o1);
  }

  // ---- Phase 3: transpose O through LDS, coalesced fp32 stores ----
#pragma unroll
  for (int i = 0; i < 8; ++i) {
    const int m = mbase + i + 8 * laneH;
    o_lds[laneN][m] = o0[i];
    o_lds[16 + laneN][m] = o1[i];
  }
  __syncthreads();

  float* obase =
      out + (((size_t)b * 256 + head * 32) * 128 + by * 8) * 128 + bx * 8;
  for (int cid = tid; cid < 256; cid += 128) {
    const int d = cid >> 3, ry = cid & 7;
    const float* srow = &o_lds[d][ry * 8];
    float* dst = obase + ((size_t)d * 128 + ry) * 128;
#pragma unroll
    for (int j = 0; j < 8; ++j) dst[j] = srow[j];
  }
}

// =====================================================================
extern "C" void kernel_launch(void* const* d_in, const int* in_sizes, int n_in,
                              void* d_out, int out_size, void* d_ws,
                              size_t ws_size, hipStream_t stream) {
  const float* x    = (const float*)d_in[0];
  const float* w_q  = (const float*)d_in[1];
  const float* w_kv = (const float*)d_in[2];
  const float* hrel = (const float*)d_in[3];
  const float* wrel = (const float*)d_in[4];
  float* out = (float*)d_out;

  const size_t q_bytes  = (size_t)32 * NBLK * 64 * DH * sizeof(__bf16);  // 32 MB
  const size_t kv_bytes = (size_t)4 * 512 * PADW * PADW * sizeof(__bf16);
  if (ws_size < q_bytes + kv_bytes) return;
  __bf16* qws = (__bf16*)d_ws;
  __bf16* kvp = (__bf16*)((char*)d_ws + q_bytes);

  (void)hipMemsetAsync(kvp, 0, kv_bytes, stream);  // zero halo borders
  halo_proj_kernel<<<dim3(4 * 12 * 128), dim3(256), 0, stream>>>(x, w_q, w_kv,
                                                                 qws, kvp);
  halo_attn_kernel<<<dim3(32 * NBLK), dim3(128), 0, stream>>>(qws, kvp, hrel,
                                                              wrel, out);
}
